// FuzzyMealy_13022340842118
// MI455X (gfx1250) — matrix operand order repaired and checked by reading, not compiled, not run on hardware
//
#include <hip/hip_runtime.h>

// Problem constants
constexpr int kS = 512;    // NUM_STATES
constexpr int kV = 64;     // NUM_INPUTS
constexpr int kK = 128;    // NUM_OUTPUTS
constexpr int kL = 8192;   // SEQ_LEN
// Scan chunking
constexpr int kG = 128;         // steps per chunk
constexpr int kC = kL / kG;     // 64 chunks
constexpr int kR = 64;          // slab rows per workgroup in phase A
constexpr int kSlabs = kS / kR; // 8 workgroups per chunk

typedef __attribute__((ext_vector_type(16))) _Float16 v16h;
typedef __attribute__((ext_vector_type(8)))  _Float16 v8h;
typedef __attribute__((ext_vector_type(8)))  float    v8f;

// ---------------------------------------------------------------------------
// Kernel 1: softmax over last axis of T_raw[s][v][:], stored TRANSPOSED as
// Tt[v][s'][s] (f16) so WMMA B-fragments are contiguous per lane.
// ---------------------------------------------------------------------------
__global__ __launch_bounds__(256)
void fm_softmaxT(const float* __restrict__ Traw, _Float16* __restrict__ Tt) {
  const int s = blockIdx.x / kV;
  const int v = blockIdx.x % kV;
  const float* row = Traw + ((size_t)s * kV + v) * kS;
  __shared__ float red[256];
  const int tid = threadIdx.x;
  float x0 = row[tid];
  float x1 = row[tid + 256];
  red[tid] = fmaxf(x0, x1);
  __syncthreads();
  for (int off = 128; off > 0; off >>= 1) {
    if (tid < off) red[tid] = fmaxf(red[tid], red[tid + off]);
    __syncthreads();
  }
  const float mx = red[0];
  __syncthreads();
  float e0 = __expf(x0 - mx), e1 = __expf(x1 - mx);
  red[tid] = e0 + e1;
  __syncthreads();
  for (int off = 128; off > 0; off >>= 1) {
    if (tid < off) red[tid] += red[tid + off];
    __syncthreads();
  }
  const float inv = 1.0f / red[0];
  _Float16* base = Tt + (size_t)v * kS * kS + s;   // Tt[v][sp][s]
  base[(size_t)tid * kS]         = (_Float16)(e0 * inv);
  base[(size_t)(tid + 256) * kS] = (_Float16)(e1 * inv);
}

// ---------------------------------------------------------------------------
// Kernel 2: softmax over last axis of O_raw[s][v][:], kept f32 row-major.
// ---------------------------------------------------------------------------
__global__ __launch_bounds__(128)
void fm_softmaxO(const float* __restrict__ Oraw, float* __restrict__ Os) {
  const int rowi = blockIdx.x;                 // (s*kV + v)
  const float* row = Oraw + (size_t)rowi * kK;
  __shared__ float red[128];
  const int tid = threadIdx.x;
  float x = row[tid];
  red[tid] = x;
  __syncthreads();
  for (int off = 64; off > 0; off >>= 1) {
    if (tid < off) red[tid] = fmaxf(red[tid], red[tid + off]);
    __syncthreads();
  }
  const float mx = red[0];
  __syncthreads();
  float e = __expf(x - mx);
  red[tid] = e;
  __syncthreads();
  for (int off = 64; off > 0; off >>= 1) {
    if (tid < off) red[tid] += red[tid + off];
    __syncthreads();
  }
  Os[(size_t)rowi * kK + tid] = e / red[0];
}

// ---------------------------------------------------------------------------
// Kernel 3 (Phase A): per-chunk transition-matrix products via WMMA.
// Each block owns an independent kR x kS row-slab of the running product and
// chains slab <- slab @ T_{v_t}. B operands (K=32 slices of T_v, 32 KB) are
// staged into LDS once per block per slice (double-buffered), shared by all
// 8 waves. Each wave register-tiles 4x4 (64x64 output): 16 WMMAs per
// 4 A-frag + 4 B-frag LDS loads. D stays in registers for the whole step,
// so a single slab buffer suffices (barrier before write-back).
//
// Fragment layouts used here (wave32; A/B in contiguous k order, not the ISA table order):
//   A 16x32 f16 : lane=(h<<4)|r holds row r; elem e -> K=16h+e
//                 => two 16B LDS loads per lane.
//   B 32x16 f16 : lane=(h<<4)|n holds col n; elem e -> K=16h+e
//                 => one contiguous 32B LDS load per lane (n-major staging).
//   D 16x16 f32 : acc[i] -> row i+8h, col = lane&15.
// ---------------------------------------------------------------------------
__global__ __launch_bounds__(256)
void fm_phaseA(const _Float16* __restrict__ Tt, const int* __restrict__ seq,
               _Float16* __restrict__ M) {
  __shared__ _Float16 slab[kR * kS];         // 64 KB running product slab
  __shared__ _Float16 bstage[2][kS * 32];    // 2 x 32 KB: Bs[n][k_local]

  const int c    = blockIdx.x / kSlabs;   // chunk
  const int j    = blockIdx.x % kSlabs;   // slab index
  const int row0 = j * kR;

  // Init slab = identity rows
  for (int idx = threadIdx.x; idx < kR * kS; idx += 256) {
    const int r = idx >> 9, col = idx & 511;
    slab[idx] = (col == row0 + r) ? (_Float16)1.0f : (_Float16)0.0f;
  }

  const int wave = threadIdx.x >> 5;
  const int lane = threadIdx.x & 31;
  const int h    = lane >> 4;
  const int lr   = lane & 15;
  const int nt0  = wave * 4;              // this wave's 4 N-tiles

  // Stage one K=32 slice of T_v into LDS: Bs[n][kl] = Tt[v][n][kk*32+kl].
  auto stageSlice = [&](const _Float16* __restrict__ Tv, int kk, int buf) {
    const _Float16* src = Tv + kk * 32;
    _Float16* dst = bstage[buf];
    // 512 rows x 64 B = 2048 chunks of 16 B
#pragma unroll
    for (int q0 = 0; q0 < 2048; q0 += 256) {
      const int q = q0 + threadIdx.x;
      const int n = q >> 2, cp = q & 3;
      *(v8h*)(dst + n * 32 + cp * 8) = *(const v8h*)(src + (size_t)n * kS + cp * 8);
    }
  };

  __syncthreads();

  for (int t = 0; t < kG; ++t) {
    const int v = seq[c * kG + t];
    const _Float16* __restrict__ Tv = Tt + (size_t)v * kS * kS;

    // Prefetch next symbol's kk=0 slice (first 64 B of each of 512 rows).
    if (t + 1 < kG) {
      const _Float16* nv = Tt + (size_t)seq[c * kG + t + 1] * kS * kS;
      __builtin_prefetch(nv + (size_t)threadIdx.x * kS, 0, 0);
      __builtin_prefetch(nv + (size_t)(threadIdx.x + 256) * kS, 0, 0);
    }

    stageSlice(Tv, 0, 0);
    __syncthreads();

    v8f acc[4][4] = {};
    for (int kk = 0; kk < 16; ++kk) {
      const int cb = kk & 1;
      if (kk + 1 < 16) stageSlice(Tv, kk + 1, cb ^ 1);

      // 4 A fragments (tile rows 0..3 of slab) for this kk
      v16h a[4];
#pragma unroll
      for (int ti = 0; ti < 4; ++ti) {
        const _Float16* p = slab + (size_t)(ti * 16 + lr) * kS + kk * 32 + 16 * h;
        v8h lo = *(const v8h*)p;          // K = kk*32 + 16h   .. +7
        v8h hi = *(const v8h*)(p + 8);    // K = kk*32 +16h+8 .. +7
        v16h f;
#pragma unroll
        for (int i = 0; i < 8; ++i) { f[i] = lo[i]; f[i + 8] = hi[i]; }
        a[ti] = f;
      }
      // 4 B fragments from staged slice
      v16h b[4];
#pragma unroll
      for (int ni = 0; ni < 4; ++ni) {
        const int n = (nt0 + ni) * 16 + lr;
        b[ni] = *(const v16h*)(bstage[cb] + n * 32 + 16 * h);
      }
#pragma unroll
      for (int ti = 0; ti < 4; ++ti)
#pragma unroll
        for (int ni = 0; ni < 4; ++ni)
          acc[ti][ni] = __builtin_amdgcn_wmma_f32_16x16x32_f16(
              false, a[ti], false, b[ni], (short)0, acc[ti][ni], false, false);

      __syncthreads();   // slice consumed; next slice staged
    }

    // All A reads done (last barrier above) -> write D back into slab (f16)
#pragma unroll
    for (int ti = 0; ti < 4; ++ti)
#pragma unroll
      for (int ni = 0; ni < 4; ++ni)
#pragma unroll
        for (int i = 0; i < 8; ++i) {
          const int row = ti * 16 + i + 8 * h;
          const int col = (nt0 + ni) * 16 + lr;
          slab[row * kS + col] = (_Float16)acc[ti][ni][i];
        }
    __syncthreads();
  }

  // Write final slab to M[c]
  _Float16* Mc = M + (size_t)c * kS * kS + (size_t)row0 * kS;
  for (int idx = threadIdx.x; idx < kR * kS; idx += 256)
    Mc[idx] = slab[idx];
}

// ---------------------------------------------------------------------------
// Kernel 4 (Phase B): serially advance boundary states through the M_c.
// Single block; 64 matvecs of 512x512 (coalesced f16 column reads).
// ---------------------------------------------------------------------------
__global__ __launch_bounds__(kS)
void fm_phaseB(const _Float16* __restrict__ M, float* __restrict__ bstates,
               float* __restrict__ states) {
  __shared__ float b[kS];
  const int n = threadIdx.x;
  const float init = (n == 0) ? 1.0f : 0.0f;
  b[n] = init;
  bstates[n] = init;
  states[n] = init;          // states[0] = one-hot
  __syncthreads();
  for (int c = 0; c < kC; ++c) {
    const _Float16* Mc = M + (size_t)c * kS * kS;
    float acc = 0.0f;
    for (int s = 0; s < kS; ++s)
      acc += b[s] * (float)Mc[(size_t)s * kS + n];
    __syncthreads();
    b[n] = acc;
    bstates[(size_t)(c + 1) * kS + n] = acc;
    __syncthreads();
  }
}

// ---------------------------------------------------------------------------
// Kernel 5 (Phase C): replay each chunk in parallel from its boundary state,
// writing every interior state (f32). Per-thread contiguous 16 B f16 streams
// from Tt; state broadcast from LDS.
// ---------------------------------------------------------------------------
__global__ __launch_bounds__(kS)
void fm_phaseC(const _Float16* __restrict__ Tt, const int* __restrict__ seq,
               const float* __restrict__ bstates, float* __restrict__ states) {
  __shared__ float st[kS];
  const int c = blockIdx.x;
  const int n = threadIdx.x;
  st[n] = bstates[(size_t)c * kS + n];
  __syncthreads();
  for (int t = 0; t < kG; ++t) {
    const int v = seq[c * kG + t];
    const _Float16* Trow = Tt + ((size_t)v * kS + n) * kS;  // Tt[v][n][*]
    float acc = 0.0f;
    for (int s = 0; s < kS; s += 8) {
      v8h tv = *(const v8h*)(Trow + s);
#pragma unroll
      for (int i = 0; i < 8; ++i) acc += st[s + i] * (float)tv[i];
    }
    __syncthreads();
    st[n] = acc;
    states[(size_t)(c * kG + t + 1) * kS + n] = acc;
    __syncthreads();
  }
}

// ---------------------------------------------------------------------------
// Kernel 6: emissions out[t] = states[t] @ O[:, v_t, :]. Fully parallel.
// ---------------------------------------------------------------------------
__global__ __launch_bounds__(kK)
void fm_outputs(const float* __restrict__ Os, const int* __restrict__ seq,
                const float* __restrict__ states, float* __restrict__ out) {
  __shared__ float st[kS];
  const int t = blockIdx.x;
  const int k = threadIdx.x;
  for (int i = k; i < kS; i += kK) st[i] = states[(size_t)t * kS + i];
  __syncthreads();
  const int v = seq[t];
  float acc = 0.0f;
  for (int s = 0; s < kS; ++s)
    acc += st[s] * Os[((size_t)s * kV + v) * kK + k];
  out[(size_t)t * kK + k] = acc;
}

// ---------------------------------------------------------------------------
// Launch
// ---------------------------------------------------------------------------
extern "C" void kernel_launch(void* const* d_in, const int* in_sizes, int n_in,
                              void* d_out, int out_size, void* d_ws, size_t ws_size,
                              hipStream_t stream) {
  (void)in_sizes; (void)n_in; (void)out_size; (void)ws_size;
  const float* Traw = (const float*)d_in[0];   // [512,64,512] f32
  const float* Oraw = (const float*)d_in[1];   // [512,64,128] f32
  const int*   seq  = (const int*)d_in[2];     // [8192] i32

  float* out    = (float*)d_out;               // [8192,128]
  float* states = out + (size_t)kL * kK;       // [8193,512]

  // Workspace layout (~80.2 MB):
  char* ws = (char*)d_ws;
  _Float16* Tt = (_Float16*)ws;                                  // 32 MB: Tt[v][s'][s]
  float*    Os = (float*)(ws + (size_t)32 * 1024 * 1024);        // 16 MB: softmax(O)
  _Float16* M  = (_Float16*)(ws + (size_t)48 * 1024 * 1024);     // 32 MB: chunk products
  float* bst   = (float*)(ws + (size_t)80 * 1024 * 1024);        // 130 KB: boundary states

  fm_softmaxT<<<kS * kV, 256, 0, stream>>>(Traw, Tt);
  fm_softmaxO<<<kS * kV, kK, 0, stream>>>(Oraw, Os);
  fm_phaseA<<<kC * kSlabs, 256, 0, stream>>>(Tt, seq, M);
  fm_phaseB<<<1, kS, 0, stream>>>(M, bst, states);
  fm_phaseC<<<kC, kS, 0, stream>>>(Tt, seq, bst, states);
  fm_outputs<<<kL, kK, 0, stream>>>(Os, seq, states, out);
}
